// MT_MoE_73237782331877
// MI455X (gfx1250) — matrix operand order repaired and checked by reading, not compiled, not run on hardware
//
#include <hip/hip_runtime.h>
#include <hip/hip_bf16.h>

// Problem constants (from reference)
constexpr int B = 8, S = 2048, D = 768, F = 3072, E = 16, CAP = 64;
constexpr float EPS = 1e-6f;

typedef __attribute__((ext_vector_type(16))) _Float16 v16h;
typedef __attribute__((ext_vector_type(8)))  _Float16 v8h;
typedef __attribute__((ext_vector_type(4)))  _Float16 v4h;
typedef __attribute__((ext_vector_type(8)))  float    v8f;

// LDS strides in halves, padded (+8) so 16-lane row accesses hit distinct banks
constexpr int XSTRIDE = D + 8;      // 776
constexpr int FCHUNK  = 128;        // F processed in chunks of 128
constexpr int HSTRIDE = FCHUNK + 8; // 136
constexpr size_t SMEM_BYTES = (size_t)(64 * XSTRIDE + 64 * HSTRIDE) * sizeof(_Float16);

// ---------------------------------------------------------------------------
// WMMA helpers
// ---------------------------------------------------------------------------
__device__ inline v8f wmma_f16(v16h a, v16h b, v8f c) {
  return __builtin_amdgcn_wmma_f32_16x16x32_f16(false, a, false, b, (short)0, c,
                                                false, false);
}

// A-matrix fragment (16x32 f16) from LDS, row-major with stride (halves).
// ISA layout: lane<16 holds row=lane, K = kb..kb+7 and kb+16..kb+23;
//             lane>=16 holds row=lane-16, K = kb+8..kb+15 and kb+24..kb+31.
__device__ inline v16h load_a_lds(const _Float16* base, int stride, int m0,
                                  int kb, int lane) {
  int row = m0 + (lane & 15);
  int k0  = kb + ((lane >> 4) << 3);
  const _Float16* p = base + row * stride + k0;
  v8h lo = *(const v8h*)p;
  v8h hi = *(const v8h*)(p + 16);
  v16h r;
#pragma unroll
  for (int i = 0; i < 8; ++i) { r[i] = lo[i]; r[i + 8] = hi[i]; }
  return r;
}

// B-matrix fragment (32x16 f16) from the pre-packed weight buffer.
// Pack layout: tile T = kt*ntiles+nt; halves for (T, lane) are 16 contiguous
// values (element j<8: K = (lane>>4)*8+j, j>=8: K = 8+(lane>>4)*8+j; col = lane&15). Two b128 loads; the
// wave reads 1 KB fully coalesced.
__device__ inline v16h load_b_packed(const _Float16* __restrict__ P, int ntiles,
                                     int kt, int nt, int lane) {
  const _Float16* q = P + ((((size_t)kt * ntiles + nt) * 32 + lane) << 4);
  v8h lo = *(const v8h*)q;
  v8h hi = *(const v8h*)(q + 8);
  v16h r;
#pragma unroll
  for (int i = 0; i < 8; ++i) { r[i] = lo[i]; r[i + 8] = hi[i]; }
  return r;
}

// ---------------------------------------------------------------------------
// Kernel 0: pack a fp32 weight set [E x Kdim x Ndim] (row-major per expert)
// into the WMMA B-fragment f16 layout described above. One-shot, ~memory-bound.
// Each thread produces the 16 contiguous halves of one (tile, lane) group.
// ---------------------------------------------------------------------------
__global__ void k_pack_w(const float* __restrict__ W, _Float16* __restrict__ P,
                         int Kdim, int Ndim) {
  const size_t ngroups = (size_t)E * Kdim * Ndim / 16;
  size_t g = (size_t)blockIdx.x * blockDim.x + threadIdx.x;
  if (g >= ngroups) return;
  const int ntiles = Ndim / 16;
  const int ktiles = Kdim / 32;
  const size_t tilesPerExp = (size_t)ktiles * ntiles;

  const int L = (int)(g % 32);
  const size_t T = g / 32;
  const int e = (int)(T / tilesPerExp);
  const size_t t = T % tilesPerExp;
  const int kt = (int)(t / ntiles), nt = (int)(t % ntiles);

  const float* We = W + (size_t)e * Kdim * Ndim;
  const int col  = nt * 16 + (L & 15);
  const int krow = kt * 32 + ((L >> 4) << 3);

  v8h lo, hi;
#pragma unroll
  for (int j = 0; j < 8; ++j) lo[j] = (_Float16)We[(size_t)(krow + j) * Ndim + col];
#pragma unroll
  for (int j = 0; j < 8; ++j) hi[j] = (_Float16)We[(size_t)(krow + 16 + j) * Ndim + col];

  _Float16* q = P + g * 16;
  *(v8h*)q       = lo;
  *(v8h*)(q + 8) = hi;
}

// ---------------------------------------------------------------------------
// Kernel 1: RMSNorm + top-1 router (one wave32 per token)
// ---------------------------------------------------------------------------
__global__ void __launch_bounds__(256)
k_norm_router(const float* __restrict__ x, const float* __restrict__ ln,
              const float* __restrict__ wr, float* __restrict__ normed,
              float* __restrict__ prob, int* __restrict__ eidx) {
  const int wave = threadIdx.x >> 5, lane = threadIdx.x & 31;
  const int t = blockIdx.x * 8 + wave;  // token id, t < B*S
  const float* xr = x + (size_t)t * D;

  float xv[D / 32];
  float ss = 0.f;
#pragma unroll
  for (int i = 0; i < D / 32; ++i) {
    float v = xr[lane + 32 * i];
    xv[i] = v;
    ss += v * v;
  }
#pragma unroll
  for (int m = 16; m >= 1; m >>= 1) ss += __shfl_xor(ss, m, 32);
  const float rstd = rsqrtf(ss / (float)D + EPS);

  float* nr = normed + (size_t)t * D;
#pragma unroll
  for (int i = 0; i < D / 32; ++i) {
    int d = lane + 32 * i;
    nr[d] = xv[i] * rstd * ln[d];
  }
  __syncthreads();  // make normed visible to all lanes for the logit pass

  // logits: lane = e + 16*half, each half covers D/2 features
  const int e = lane & 15, half = lane >> 4;
  float acc = 0.f;
  for (int d = half * (D / 2); d < (half + 1) * (D / 2); ++d)
    acc += nr[d] * wr[d * E + e];
  acc += __shfl_xor(acc, 16, 32);  // both halves now hold logit[e]

  float mx = acc;
#pragma unroll
  for (int m = 8; m >= 1; m >>= 1) mx = fmaxf(mx, __shfl_xor(mx, m, 32));
  float p = __expf(acc - mx);
  float sum = p;
#pragma unroll
  for (int m = 8; m >= 1; m >>= 1) sum += __shfl_xor(sum, m, 32);
  // argmax with first-index tie-break
  int cand = (acc == mx) ? e : E;
#pragma unroll
  for (int m = 8; m >= 1; m >>= 1) cand = min(cand, __shfl_xor(cand, m, 32));

  if (lane == 0) {
    prob[t] = 1.0f / sum;  // softmax prob of the max logit
    eidx[t] = cand;
  }
}

// ---------------------------------------------------------------------------
// Kernel 2: capacity assignment (sequential per (b,e), matches cumsum order)
// ---------------------------------------------------------------------------
__global__ void k_capacity(const int* __restrict__ eidx, int* __restrict__ counts,
                           int* __restrict__ slots) {
  int tid = blockIdx.x * blockDim.x + threadIdx.x;
  if (tid >= B * E) return;
  const int b = tid / E, e = tid % E;
  const int* row = eidx + (size_t)b * S;
  int cnt = 0;
  for (int s = 0; s < S; ++s) {
    if (row[s] == e) {
      if (cnt < CAP) slots[tid * CAP + cnt] = s;
      ++cnt;
    }
  }
  int kept = cnt < CAP ? cnt : CAP;
  counts[tid] = kept;
  for (int c = kept; c < CAP; ++c) slots[tid * CAP + c] = -1;
}

// ---------------------------------------------------------------------------
// Kernel 3: pass-through for every token: out = x + prob * normed
// (routed tokens get overwritten by the expert kernel afterwards)
// ---------------------------------------------------------------------------
__global__ void k_pass(const float* __restrict__ x, const float* __restrict__ normed,
                       const float* __restrict__ prob, float* __restrict__ out) {
  size_t i = (size_t)blockIdx.x * blockDim.x + threadIdx.x;  // float4 index
  if (i >= (size_t)B * S * D / 4) return;
  int t = (int)(i / (D / 4));
  float p = prob[t];
  float4 a = ((const float4*)x)[i];
  float4 n = ((const float4*)normed)[i];
  float4 r;
  r.x = a.x + p * n.x; r.y = a.y + p * n.y;
  r.z = a.z + p * n.z; r.w = a.w + p * n.w;
  ((float4*)out)[i] = r;
}

// ---------------------------------------------------------------------------
// Kernel 4: fused expert FFN per (b,e): gather -> WMMA GEMM1 -> ReLU ->
//           WMMA GEMM2 -> combine+residual scatter. 512 threads = 16 wave32.
// ---------------------------------------------------------------------------
__global__ void __launch_bounds__(512, 1)
k_expert(const float* __restrict__ x, const float* __restrict__ normed,
         const _Float16* __restrict__ wiP, const _Float16* __restrict__ woP,
         const float* __restrict__ prob, const int* __restrict__ counts,
         const int* __restrict__ slots, float* __restrict__ out) {
  extern __shared__ __align__(16) char smem[];
  _Float16* Xa = (_Float16*)smem;                       // [64 x XSTRIDE]
  _Float16* Hc = Xa + 64 * XSTRIDE;                     // [64 x HSTRIDE]

  const int b = blockIdx.x / E, e = blockIdx.x % E;
  const int be = b * E + e;
  const int cnt = counts[be];
  if (cnt == 0) return;  // uniform: whole block exits, no WMMA issued

  const int tid = threadIdx.x;
  const int wave = tid >> 5, lane = tid & 31;

  // --- Gather up to 64 token rows of normed into LDS as f16 ---
  {
    const int r = tid >> 3;          // 0..63, 8 threads per row
    const int seg = tid & 7;         // 96 floats per thread
    int s = slots[be * CAP + r];
    const bool keep = (r < cnt) && (s >= 0);
    const float* src = normed + ((size_t)b * S + (keep ? s : 0)) * D;
#pragma unroll
    for (int i = seg * 96; i < seg * 96 + 96; i += 4) {
      float4 v = keep ? *(const float4*)(src + i)
                      : make_float4(0.f, 0.f, 0.f, 0.f);
      v4h h;
      h[0] = (_Float16)v.x; h[1] = (_Float16)v.y;
      h[2] = (_Float16)v.z; h[3] = (_Float16)v.w;
      *(v4h*)(&Xa[r * XSTRIDE + i]) = h;
    }
  }
  __syncthreads();

  // Wave tiling: wm in 0..3 -> M tile (16 rows), wg in 0..3 -> 192-col slab
  const int wm = wave & 3;
  const int wg = wave >> 2;
  const int nbase = wg * 192;        // 12 N-tiles of 16 for GEMM2 output

  v8f acc[12];
#pragma unroll
  for (int t = 0; t < 12; ++t) acc[t] = (v8f){};

  constexpr int NT_WI = F / 16;   // 192 n-tiles in packed wi
  constexpr int NT_WO = D / 16;   // 48  n-tiles in packed wo
  const _Float16* wi_e = wiP + (size_t)e * D * F;   // packed, per-expert
  const _Float16* wo_e = woP + (size_t)e * F * D;   // packed, per-expert

  for (int fc = 0; fc < F; fc += FCHUNK) {
    // Prefetch next chunk's wo slab (K rows fc+FCHUNK..) while GEMM1 runs.
    if (fc + FCHUNK < F) {
      const _Float16* nxt =
          wo_e + (((size_t)((fc + FCHUNK) / 32) * NT_WO) * 32 + tid) * 16;
      __builtin_prefetch(nxt, 0, 1);
    }

    // ---- GEMM1: Hc[64 x 128] = ReLU(Xa[64 x 768] @ wi[:, fc:fc+128]) ----
    // Wave computes tiles (wm, wg*2) and (wm, wg*2+1)
    v8f h0 = (v8f){}, h1 = (v8f){};
    const int nt0 = (fc >> 4) + wg * 2;
    for (int kb = 0; kb < D; kb += 32) {
      v16h a  = load_a_lds(Xa, XSTRIDE, wm * 16, kb, lane);
      v16h b0 = load_b_packed(wi_e, NT_WI, kb >> 5, nt0, lane);
      v16h b1 = load_b_packed(wi_e, NT_WI, kb >> 5, nt0 + 1, lane);
      h0 = wmma_f16(a, b0, h0);
      h1 = wmma_f16(a, b1, h1);
    }
    __syncthreads();  // previous chunk's Hc fully consumed

    // C-layout store with ReLU + f16 convert:
    // vgpr r: lane<16 -> M=r, N=lane ; lane>=16 -> M=8+r, N=lane-16
    {
      const int row0 = wm * 16 + ((lane >> 4) << 3);
      const int col0 = (wg * 2) * 16 + (lane & 15);
#pragma unroll
      for (int r = 0; r < 8; ++r) {
        float a0 = h0[r]; a0 = a0 > 0.f ? a0 : 0.f;
        float a1 = h1[r]; a1 = a1 > 0.f ? a1 : 0.f;
        Hc[(row0 + r) * HSTRIDE + col0]      = (_Float16)a0;
        Hc[(row0 + r) * HSTRIDE + col0 + 16] = (_Float16)a1;
      }
    }
    __syncthreads();  // Hc ready for all waves

    // ---- GEMM2: acc[64 x 768] += Hc[64 x 128] @ wo[fc:fc+128, :] ----
    v16h afr[4];
#pragma unroll
    for (int kk = 0; kk < 4; ++kk)
      afr[kk] = load_a_lds(Hc, HSTRIDE, wm * 16, kk * 32, lane);
#pragma unroll
    for (int t = 0; t < 12; ++t) {
      const int nt = (nbase >> 4) + t;
#pragma unroll
      for (int kk = 0; kk < 4; ++kk) {
        v16h bb = load_b_packed(wo_e, NT_WO, (fc >> 5) + kk, nt, lane);
        acc[t] = wmma_f16(afr[kk], bb, acc[t]);
      }
    }
  }

  // ---- Epilogue: out[b,s,:] = x[b,s,:] + prob[b,s] * expert_out ----
  {
    const int rowA = (lane >> 4) << 3;
    const int col  = lane & 15;
    int   sTok[8];
    float pv[8];
#pragma unroll
    for (int r = 0; r < 8; ++r) {
      int row = wm * 16 + rowA + r;
      bool v = row < cnt;
      int s = v ? slots[be * CAP + row] : -1;
      sTok[r] = s;
      pv[r] = (v && s >= 0) ? prob[b * S + s] : 0.f;
    }
#pragma unroll
    for (int t = 0; t < 12; ++t) {
      const int n = nbase + t * 16 + col;
#pragma unroll
      for (int r = 0; r < 8; ++r) {
        if (sTok[r] >= 0) {
          size_t o = ((size_t)b * S + sTok[r]) * D + n;
          out[o] = x[o] + pv[r] * acc[t][r];
        }
      }
    }
  }
}

// ---------------------------------------------------------------------------
extern "C" void kernel_launch(void* const* d_in, const int* in_sizes, int n_in,
                              void* d_out, int out_size, void* d_ws, size_t ws_size,
                              hipStream_t stream) {
  const float* x  = (const float*)d_in[0];
  const float* ln = (const float*)d_in[1];
  const float* wr = (const float*)d_in[2];
  const float* wi = (const float*)d_in[3];
  const float* wo = (const float*)d_in[4];
  float* out = (float*)d_out;

  char* ws = (char*)d_ws;
  float*     normed = (float*)ws;     ws += (size_t)B * S * D * sizeof(float);
  float*     prob   = (float*)ws;     ws += (size_t)B * S * sizeof(float);
  int*       eidx   = (int*)ws;       ws += (size_t)B * S * sizeof(int);
  int*       counts = (int*)ws;       ws += (size_t)B * E * sizeof(int);
  int*       slots  = (int*)ws;       ws += (size_t)B * E * CAP * sizeof(int);
  _Float16*  wiP    = (_Float16*)ws;  ws += (size_t)E * D * F * sizeof(_Float16);
  _Float16*  woP    = (_Float16*)ws;  ws += (size_t)E * F * D * sizeof(_Float16);

  // Weight pre-pack into WMMA B-fragment f16 layout (one-shot, memory bound)
  {
    size_t ngroups = (size_t)E * D * F / 16;
    int blocks = (int)((ngroups + 255) / 256);
    k_pack_w<<<blocks, 256, 0, stream>>>(wi, wiP, D, F);
    k_pack_w<<<blocks, 256, 0, stream>>>(wo, woP, F, D);
  }

  k_norm_router<<<B * S / 8, 256, 0, stream>>>(x, ln, wr, normed, prob, eidx);
  k_capacity<<<(B * E + 63) / 64, 64, 0, stream>>>(eidx, counts, slots);
  k_pass<<<(int)(((size_t)B * S * D / 4 + 255) / 256), 256, 0, stream>>>(
      x, normed, prob, out);
  k_expert<<<B * E, 512, SMEM_BYTES, stream>>>(x, normed, wiP, woP, prob, counts,
                                               slots, out);
}
